// PhiMoeSparseMoeBlock_28252294873715
// MI455X (gfx1250) — matrix operand order repaired and checked by reading, not compiled, not run on hardware
//
#include <hip/hip_runtime.h>

// ---------------- problem constants ----------------
#define T_TOK 2048
#define H_DIM 1024
#define F_DIM 4096
#define E_NUM 8
#define MAXROWS 5120            // sum_e ceil(cnt_e/128)*128 <= 2T + 8*127 = 5112
#define MTILES (MAXROWS / 128)  // 40
#define BROW 40                 // LDS B row stride in bf16 elems (80B, 16B-aligned)
#define BTILE (64 * BROW)       // one B buffer (bf16 elems)

typedef __bf16 v16bf __attribute__((ext_vector_type(16)));
typedef __bf16 bf16x8 __attribute__((ext_vector_type(8)));
typedef float  v8f   __attribute__((ext_vector_type(8)));

__device__ __forceinline__ unsigned short f2bfbits(float f) {      // RNE (cold paths)
    unsigned u = __builtin_bit_cast(unsigned, f);
    return (unsigned short)((u + 0x7FFFu + ((u >> 16) & 1u)) >> 16);
}
__device__ __forceinline__ unsigned short f2bfbits_trunc(float f) { // hot staging path:
    return (unsigned short)(__builtin_bit_cast(unsigned, f) >> 16); // folds to d16_hi store
}
__device__ __forceinline__ float fastrcp(float x) { return __builtin_amdgcn_rcpf(x); }

// ---------------- 1) X f32 -> bf16 ----------------
__global__ __launch_bounds__(256) void xcast_kernel(const float* __restrict__ x,
                                                    unsigned short* __restrict__ xb) {
    size_t i = ((size_t)blockIdx.x * 256 + threadIdx.x) * 4;
    float4 v = *(const float4*)(x + i);
    ushort4 o;
    o.x = f2bfbits(v.x); o.y = f2bfbits(v.y);
    o.z = f2bfbits(v.z); o.w = f2bfbits(v.w);
    *(ushort4*)(xb + i) = o;
}

// ---------------- 2) router + sparsemixer + gather lists ----------------
__global__ __launch_bounds__(256) void router_kernel(const float* __restrict__ x,
                                                     const float* __restrict__ gw,
                                                     float* __restrict__ logits_out,
                                                     int* __restrict__ routedTok,
                                                     float* __restrict__ routedW,
                                                     int* __restrict__ counts) {
    __shared__ float sgw[H_DIM * E_NUM];   // 32 KB
    __shared__ float slog[32][E_NUM];
    for (int i = threadIdx.x; i < H_DIM * E_NUM / 4; i += 256)
        ((float4*)sgw)[i] = ((const float4*)gw)[i];
    __syncthreads();

    int lt  = threadIdx.x >> 3;      // token within block (0..31)
    int e   = threadIdx.x & 7;       // expert
    int tok = blockIdx.x * 32 + lt;
    const float* xr = x + (size_t)tok * H_DIM;
    float acc = 0.f;
    for (int h = 0; h < H_DIM; h += 4) {
        float4 xv = *(const float4*)(xr + h);
        acc += xv.x * sgw[(h + 0) * E_NUM + e] + xv.y * sgw[(h + 1) * E_NUM + e]
             + xv.z * sgw[(h + 2) * E_NUM + e] + xv.w * sgw[(h + 3) * E_NUM + e];
    }
    slog[lt][e] = acc;
    logits_out[(size_t)tok * E_NUM + e] = acc;
    __syncthreads();

    if (e == 0) {
        float s[E_NUM];
        for (int j = 0; j < E_NUM; ++j) s[j] = slog[lt][j];
        const float eps2 = 0.02f;  // 2 * jitter_eps
        // top-1 (first max index, like jnp.argmax)
        int i1 = 0; float m1 = s[0];
        for (int j = 1; j < E_NUM; ++j) if (s[j] > m1) { m1 = s[j]; i1 = j; }
        float sum1 = 0.f;
        for (int j = 0; j < E_NUM; ++j) {
            float fac = fmaxf(fabsf(s[j]), m1);
            // (m1-s)/fac > eps2  <=>  (m1-s) > eps2*fac  (fac>=0; fac==0 => both keep)
            if ((m1 - s[j]) > eps2 * fac) continue;
            sum1 += __expf(s[j] - m1);
        }
        float mult1 = fastrcp(sum1);                  // exp(0)/sum
        // top-2 over scores with i1 masked out
        int i2 = 0; float m2 = -3.4e38f; bool first = true;
        for (int j = 0; j < E_NUM; ++j) {
            if (j == i1) continue;
            if (first || s[j] > m2) { m2 = s[j]; i2 = j; first = false; }
        }
        float sum2 = 0.f;
        for (int j = 0; j < E_NUM; ++j) {
            if (j == i1) continue;                    // masked_scores[i1] = -inf
            float fac = fmaxf(fabsf(s[j]), m2);
            if ((m2 - s[j]) > eps2 * fac) continue;
            sum2 += __expf(s[j] - m2);
        }
        float mult2 = fastrcp(sum2);
        // faithful quirk: weight for expert i_k is dense[t, k] (column = slot index)
        float d0 = (i1 == 0) ? mult1 : ((i2 == 0) ? mult2 : 0.f);
        float d1 = (i1 == 1) ? mult1 : ((i2 == 1) ? mult2 : 0.f);
        int p1 = atomicAdd(&counts[i1], 1);
        routedTok[i1 * T_TOK + p1] = tok;  routedW[i1 * T_TOK + p1] = d0;
        int p2 = atomicAdd(&counts[i2], 1);
        routedTok[i2 * T_TOK + p2] = tok;  routedW[i2 * T_TOK + p2] = d1;
    }
}

// ---------------- 3) 128-aligned prefix offsets ----------------
__global__ void scan_kernel(const int* __restrict__ counts, int* __restrict__ offPad) {
    if (threadIdx.x == 0) {
        int o = 0; offPad[0] = 0;
        for (int e = 0; e < E_NUM; ++e) {
            o += ((counts[e] + 127) / 128) * 128;
            offPad[e + 1] = o;
        }
    }
}

// ---------------- fragment / staging helpers ----------------
__device__ __forceinline__ v16bf load_a_frag(const unsigned short* p) {
    // per-lane 16 contiguous bf16: two 16B chunks at +0 and +8 elems (A 16x32 layout)
    union { v16bf v; bf16x8 h[2]; } U;
    const bf16x8* q = (const bf16x8*)p;
    U.h[0] = q[0]; U.h[1] = q[1];
    return U.v;
}
__device__ __forceinline__ v16bf load_b_frag(const unsigned short* p) {
    // per-lane 16 contiguous bf16 in LDS [N][K] layout (B 32x16 layout)
    union { v16bf v; bf16x8 h[2]; } U;
    const bf16x8* q = (const bf16x8*)p;
    U.h[0] = q[0]; U.h[1] = q[1];
    return U.v;
}
__device__ __forceinline__ void load8(const float* __restrict__ p, float* v) {
    float4 a0 = *(const float4*)p, a1 = *(const float4*)(p + 4);
    v[0] = a0.x; v[1] = a0.y; v[2] = a0.z; v[3] = a0.w;
    v[4] = a1.x; v[5] = a1.y; v[6] = a1.z; v[7] = a1.w;
}

// ---------------- 4) expert FFN stage 1: G = silu(X W1) * (X W3), bf16 ----------------
__global__ __launch_bounds__(256) void ffn1_kernel(const unsigned short* __restrict__ xb,
                                                   const float* __restrict__ w1,
                                                   const float* __restrict__ w3,
                                                   const int* __restrict__ routedTok,
                                                   const int* __restrict__ counts,
                                                   const int* __restrict__ offPad,
                                                   unsigned short* __restrict__ G) {
    __shared__ int sOff[E_NUM + 1];
    __shared__ int sTok[128];
    __shared__ unsigned short sB1[2 * BTILE];   // double-buffered
    __shared__ unsigned short sB3[2 * BTILE];

    if (threadIdx.x < E_NUM + 1) sOff[threadIdx.x] = offPad[threadIdx.x];
    __syncthreads();
    int mbase = blockIdx.y * 128;
    if (mbase >= sOff[E_NUM]) return;             // uniform over block
    int e = 0;
    while (e < E_NUM - 1 && mbase >= sOff[e + 1]) ++e;
    int cnt   = counts[e];
    int rbase = mbase - sOff[e];
    if (threadIdx.x < 128) {
        int r = rbase + (int)threadIdx.x;
        sTok[threadIdx.x] = (r < cnt) ? routedTok[e * T_TOK + r] : 0;
    }
    __syncthreads();

    int lane = threadIdx.x & 31, wave = threadIdx.x >> 5;
    int wm = wave >> 1, wn = wave & 1;            // 4x2 wave grid -> 128x64 tile
    int lane16 = lane & 15, laneHi = lane >> 4;
    int mrow0 = wm * 32, ncol0 = wn * 32;

    const unsigned short* aBase[2] = {
        xb + (size_t)sTok[mrow0 + lane16] * H_DIM,
        xb + (size_t)sTok[mrow0 + 16 + lane16] * H_DIM };

    const float* W1e = w1 + (size_t)e * H_DIM * F_DIM + (size_t)blockIdx.x * 64;
    const float* W3e = w3 + (size_t)e * H_DIM * F_DIM + (size_t)blockIdx.x * 64;

    v8f z = {0, 0, 0, 0, 0, 0, 0, 0};
    v8f acc1[2][2] = {{z, z}, {z, z}};
    v8f acc3[2][2] = {{z, z}, {z, z}};

    int kRow = threadIdx.x >> 3;                  // 0..31
    int n8   = (threadIdx.x & 7) * 8;             // 0..56

    const int NK = H_DIM / 32;                    // 32 K-steps
    float va[8], vb[8];
    load8(W1e + (size_t)kRow * F_DIM + n8, va);
    load8(W3e + (size_t)kRow * F_DIM + n8, vb);

    for (int kt = 0; kt < NK; ++kt) {
        unsigned short* b1 = sB1 + (kt & 1) * BTILE;
        unsigned short* b3 = sB3 + (kt & 1) * BTILE;
#pragma unroll
        for (int j = 0; j < 8; ++j) {             // truncating cast -> ds_store_b16_d16_hi
            b1[(n8 + j) * BROW + kRow] = f2bfbits_trunc(va[j]);
            b3[(n8 + j) * BROW + kRow] = f2bfbits_trunc(vb[j]);
        }
        __syncthreads();
        if (kt + 1 < NK) {   // prefetch next W tile into registers (overlaps WMMA)
            load8(W1e + (size_t)((kt + 1) * 32 + kRow) * F_DIM + n8, va);
            load8(W3e + (size_t)((kt + 1) * 32 + kRow) * F_DIM + n8, vb);
        }
        int kk = kt * 32;

        v16bf afr[2], b1f[2], b3f[2];
#pragma unroll
        for (int mf = 0; mf < 2; ++mf)
            afr[mf] = load_a_frag(aBase[mf] + kk + laneHi * 16);
#pragma unroll
        for (int nf = 0; nf < 2; ++nf) {
            int idx = (ncol0 + nf * 16 + lane16) * BROW + laneHi * 16;
            b1f[nf] = load_b_frag(&b1[idx]);
            b3f[nf] = load_b_frag(&b3[idx]);
        }
#pragma unroll
        for (int mf = 0; mf < 2; ++mf)
#pragma unroll
            for (int nf = 0; nf < 2; ++nf) {
                acc1[mf][nf] = __builtin_amdgcn_wmma_f32_16x16x32_bf16(
                    false, afr[mf], false, b1f[nf], (short)0, acc1[mf][nf], false, false);
                acc3[mf][nf] = __builtin_amdgcn_wmma_f32_16x16x32_bf16(
                    false, afr[mf], false, b3f[nf], (short)0, acc3[mf][nf], false, false);
            }
    }

    // fused SwiGLU epilogue -> bf16 G  (sigmoid via v_rcp_f32, not IEEE div)
#pragma unroll
    for (int mf = 0; mf < 2; ++mf)
#pragma unroll
        for (int nf = 0; nf < 2; ++nf) {
            int col = blockIdx.x * 64 + ncol0 + nf * 16 + lane16;
#pragma unroll
            for (int v = 0; v < 8; ++v) {
                float g = acc1[mf][nf][v];
                float u = acc3[mf][nf][v];
                float sv = g * fastrcp(1.f + __expf(-g)) * u;
                int row = mbase + mrow0 + mf * 16 + laneHi * 8 + v;
                G[(size_t)row * F_DIM + col] = f2bfbits(sv);
            }
        }
}

// ---------------- 5) expert FFN stage 2: out += w * (G W2), scatter ----------------
__global__ __launch_bounds__(256) void ffn2_kernel(const unsigned short* __restrict__ G,
                                                   const float* __restrict__ w2,
                                                   const int* __restrict__ routedTok,
                                                   const float* __restrict__ routedW,
                                                   const int* __restrict__ counts,
                                                   const int* __restrict__ offPad,
                                                   float* __restrict__ outF) {
    __shared__ int   sOff[E_NUM + 1];
    __shared__ int   sTok[128];
    __shared__ float sWgt[128];
    __shared__ unsigned short sB[2 * BTILE];    // double-buffered

    if (threadIdx.x < E_NUM + 1) sOff[threadIdx.x] = offPad[threadIdx.x];
    __syncthreads();
    int mbase = blockIdx.y * 128;
    if (mbase >= sOff[E_NUM]) return;
    int e = 0;
    while (e < E_NUM - 1 && mbase >= sOff[e + 1]) ++e;
    int cnt   = counts[e];
    int rbase = mbase - sOff[e];
    if (threadIdx.x < 128) {
        int r = rbase + (int)threadIdx.x;
        sTok[threadIdx.x] = (r < cnt) ? routedTok[e * T_TOK + r] : 0;
        sWgt[threadIdx.x] = (r < cnt) ? routedW[e * T_TOK + r] : 0.f;
    }
    __syncthreads();

    int lane = threadIdx.x & 31, wave = threadIdx.x >> 5;
    int wm = wave >> 1, wn = wave & 1;
    int lane16 = lane & 15, laneHi = lane >> 4;
    int mrow0 = wm * 32, ncol0 = wn * 32;

    const unsigned short* aBase[2] = {
        G + (size_t)(mbase + mrow0 + lane16) * F_DIM,
        G + (size_t)(mbase + mrow0 + 16 + lane16) * F_DIM };

    const float* W2e = w2 + (size_t)e * F_DIM * H_DIM + (size_t)blockIdx.x * 64;

    v8f z = {0, 0, 0, 0, 0, 0, 0, 0};
    v8f acc[2][2] = {{z, z}, {z, z}};

    int kRow = threadIdx.x >> 3;
    int n8   = (threadIdx.x & 7) * 8;

    const int NK = F_DIM / 32;                    // 128 K-steps
    float va[8];
    load8(W2e + (size_t)kRow * H_DIM + n8, va);

    for (int kt = 0; kt < NK; ++kt) {
        unsigned short* b = sB + (kt & 1) * BTILE;
#pragma unroll
        for (int j = 0; j < 8; ++j)               // truncating cast -> ds_store_b16_d16_hi
            b[(n8 + j) * BROW + kRow] = f2bfbits_trunc(va[j]);
        __syncthreads();
        if (kt + 1 < NK) {
            load8(W2e + (size_t)((kt + 1) * 32 + kRow) * H_DIM + n8, va);
        }
        int kk = kt * 32;

        v16bf afr[2], bfr[2];
#pragma unroll
        for (int mf = 0; mf < 2; ++mf)
            afr[mf] = load_a_frag(aBase[mf] + kk + laneHi * 16);
#pragma unroll
        for (int nf = 0; nf < 2; ++nf)
            bfr[nf] = load_b_frag(&b[(ncol0 + nf * 16 + lane16) * BROW + laneHi * 16]);
#pragma unroll
        for (int mf = 0; mf < 2; ++mf)
#pragma unroll
            for (int nf = 0; nf < 2; ++nf)
                acc[mf][nf] = __builtin_amdgcn_wmma_f32_16x16x32_bf16(
                    false, afr[mf], false, bfr[nf], (short)0, acc[mf][nf], false, false);
    }

    // weighted scatter-accumulate (each (tok,col) gets exactly 2 adds -> deterministic)
#pragma unroll
    for (int mf = 0; mf < 2; ++mf)
#pragma unroll
        for (int nf = 0; nf < 2; ++nf) {
            int col = blockIdx.x * 64 + ncol0 + nf * 16 + lane16;
#pragma unroll
            for (int v = 0; v < 8; ++v) {
                int rl  = mrow0 + mf * 16 + laneHi * 8 + v;
                int tok = sTok[rl];
                float wv = sWgt[rl];
                atomicAdd(&outF[(size_t)tok * H_DIM + col], acc[mf][nf][v] * wv);
            }
        }
}

// ---------------- workspace layout (bytes) ----------------
#define XB_OFF   ((size_t)0)
#define XB_BYTES ((size_t)T_TOK * H_DIM * 2)
#define G_OFF    (XB_OFF + XB_BYTES)
#define G_BYTES  ((size_t)MAXROWS * F_DIM * 2)
#define RT_OFF   (G_OFF + G_BYTES)
#define RT_BYTES ((size_t)E_NUM * T_TOK * 4)
#define RW_OFF   (RT_OFF + RT_BYTES)
#define RW_BYTES ((size_t)E_NUM * T_TOK * 4)
#define CNT_OFF  (RW_OFF + RW_BYTES)
#define OFP_OFF  (CNT_OFF + 64)

extern "C" void kernel_launch(void* const* d_in, const int* in_sizes, int n_in,
                              void* d_out, int out_size, void* d_ws, size_t ws_size,
                              hipStream_t stream) {
    const float* x  = (const float*)d_in[0];
    const float* gw = (const float*)d_in[1];
    const float* w1 = (const float*)d_in[2];
    const float* w3 = (const float*)d_in[3];
    const float* w2 = (const float*)d_in[4];

    float* outF      = (float*)d_out;
    float* outLogits = outF + (size_t)T_TOK * H_DIM;

    char* ws = (char*)d_ws;
    unsigned short* xb        = (unsigned short*)(ws + XB_OFF);
    unsigned short* G         = (unsigned short*)(ws + G_OFF);
    int*            routedTok = (int*)(ws + RT_OFF);
    float*          routedW   = (float*)(ws + RW_OFF);
    int*            counts    = (int*)(ws + CNT_OFF);
    int*            offPad    = (int*)(ws + OFP_OFF);

    hipMemsetAsync(counts, 0, E_NUM * sizeof(int), stream);
    hipMemsetAsync(outF, 0, (size_t)T_TOK * H_DIM * sizeof(float), stream);

    xcast_kernel<<<(T_TOK * H_DIM) / 1024, 256, 0, stream>>>(x, xb);
    router_kernel<<<T_TOK / 32, 256, 0, stream>>>(x, gw, outLogits, routedTok, routedW, counts);
    scan_kernel<<<1, 32, 0, stream>>>(counts, offPad);
    ffn1_kernel<<<dim3(F_DIM / 64, MTILES), 256, 0, stream>>>(xb, w1, w3, routedTok, counts, offPad, G);
    ffn2_kernel<<<dim3(H_DIM / 64, MTILES), 256, 0, stream>>>(G, w2, routedTok, routedW, counts, offPad, outF);
}
